// SparseCoreAttention_65953517797444
// MI455X (gfx1250) — compile-verified, matrix operand order checked by reading, not run on hardware
//
#include <hip/hip_runtime.h>

typedef __bf16 v16bf __attribute__((ext_vector_type(16)));
typedef float  v8f   __attribute__((ext_vector_type(8)));
typedef float  f32x4 __attribute__((ext_vector_type(4)));
typedef int    i32x2 __attribute__((ext_vector_type(2)));

constexpr int BB = 2, HH = 12, NN = 2048, DD = 128, WW = 64;
constexpr int WAVES = 4;                 // waves (queries) per workgroup
constexpr int TOTALQ = BB * HH * NN;     // 49152 independent query rows

__global__ __launch_bounds__(WAVES * 32, 1)
void sparse_attn_wmma_kernel(const float* __restrict__ q,
                             const float* __restrict__ k,
                             const float* __restrict__ v,
                             const int*   __restrict__ cols,
                             float* __restrict__ out)
{
    __shared__ float s_q[WAVES][DD];   // pre-scaled query row per wave
    __shared__ int   s_c[WAVES][WW];   // gather indices per wave

    const int lane = threadIdx.x & 31;
    const int wv   = threadIdx.x >> 5;
    const int gq   = blockIdx.x * WAVES + wv;  // query id in [0, B*H*N)
    const int bh   = gq / NN;                  // fused (b*H + h)
    const int n    = gq - bh * NN;

    const int m  = lane & 15;   // tile row / output column index
    const int hh = lane >> 4;   // lane half (VGPR-layout selector)
    const float scale = 0.08838834764831845f;  // 1/sqrt(128)

    const size_t rowoff = ((size_t)bh * NN + n) * DD;
    const size_t bhbase = (size_t)bh * NN;

    // ---- stage q (pre-scaled) + col_ids into LDS; prefetch gathered rows ----
    {
        const f32x4* q4 = (const f32x4*)(q + rowoff);
        f32x4 t = q4[lane];
        *(f32x4*)&s_q[wv][lane * 4] = t * scale;

        i32x2 cc = ((const i32x2*)(cols + (size_t)n * WW))[lane];
        s_c[wv][2 * lane + 0] = cc[0];
        s_c[wv][2 * lane + 1] = cc[1];

        // warm L2 for the rows we are about to gather (global_prefetch_b8)
        __builtin_prefetch(k + (bhbase + cc[0]) * DD, 0, 1);
        __builtin_prefetch(k + (bhbase + cc[1]) * DD, 0, 1);
        __builtin_prefetch(v + (bhbase + cc[0]) * DD, 0, 1);
        __builtin_prefetch(v + (bhbase + cc[1]) * DD, 0, 1);
    }
    __syncthreads();

    // =====================  scores = (Kg . q) * scale  =====================
    // Per w-chunk c: A = 16 gathered K rows x 32 d (bf16, WMMA A layout),
    // B = q chunk replicated across all 16 columns. D[i,j] = score[16c+i] for
    // every j, so lane-half hh ends up holding score w = 16c + r + 8*hh in
    // accumulator register r.
    float sc[32];
    #pragma unroll
    for (int c = 0; c < 4; ++c) {
        const int    krow = s_c[wv][16 * c + m];
        const f32x4* k4   = (const f32x4*)(k + (bhbase + krow) * DD);
        v8f acc = {};
        #pragma unroll
        for (int dc = 0; dc < 4; ++dc) {
            const int s0 = 8 * dc + 2 * hh;          // float4 index of segment 0
            f32x4 a0 = k4[s0],     a1 = k4[s0 + 1];  // d = 32dc+8h   .. +7
            f32x4 b0 = k4[s0 + 4], b1 = k4[s0 + 5];  // d = 32dc+16+8h.. +7
            v16bf A;
            const f32x4* qh = (const f32x4*)&s_q[wv][32 * dc + 8 * hh];
            f32x4 c0 = qh[0], c1 = qh[1], d0 = qh[4], d1 = qh[5];
            v16bf Bm;
            #pragma unroll
            for (int t = 0; t < 4; ++t) {
                A[t]      = (__bf16)a0[t];  A[4 + t]  = (__bf16)a1[t];
                A[8 + t]  = (__bf16)b0[t];  A[12 + t] = (__bf16)b1[t];
                Bm[t]     = (__bf16)c0[t];  Bm[4 + t]  = (__bf16)c1[t];
                Bm[8 + t] = (__bf16)d0[t];  Bm[12 + t] = (__bf16)d1[t];
            }
            acc = __builtin_amdgcn_wmma_f32_16x16x32_bf16(
                      false, A, false, Bm, (short)0, acc, false, false);
        }
        #pragma unroll
        for (int r = 0; r < 8; ++r) sc[8 * c + r] = acc[r];
    }

    // =====================  softmax over the 64 scores  =====================
    float mx = -3.402823466e+38f;
    #pragma unroll
    for (int i = 0; i < 32; ++i) mx = fmaxf(mx, sc[i]);
    mx = fmaxf(mx, __shfl_xor(mx, 16, 32));       // combine lane halves
    float sum = 0.0f;
    #pragma unroll
    for (int i = 0; i < 32; ++i) { float e = __expf(sc[i] - mx); sc[i] = e; sum += e; }
    sum += __shfl_xor(sum, 16, 32);
    const float inv = 1.0f / sum;
    #pragma unroll
    for (int i = 0; i < 32; ++i) sc[i] *= inv;    // sc[] now holds probs

    // =====================  out = probs . Vg  =====================
    // A = probs chunk replicated across all 16 rows; B = gathered V columns
    // (dword gathers, coalesced 64B per half-wave). D[i,j] = out[16*odc + j].
    int vcol[32];
    #pragma unroll
    for (int wc = 0; wc < 2; ++wc)
        #pragma unroll
        for (int e = 0; e < 16; ++e) {
            const int w = 32 * wc + 8 * hh + e + (e >= 8 ? 8 : 0);
            vcol[16 * wc + e] = s_c[wv][w];
        }

    #pragma unroll
    for (int odc = 0; odc < 8; ++odc) {
        const int d = 16 * odc + m;
        v8f acc = {};
        #pragma unroll
        for (int wc = 0; wc < 2; ++wc) {
            v16bf A, Bv;
            #pragma unroll
            for (int e = 0; e < 16; ++e) {
                A[e]  = (__bf16)sc[16 * wc + e];
                Bv[e] = (__bf16)v[(bhbase + vcol[16 * wc + e]) * (size_t)DD + d];
            }
            acc = __builtin_amdgcn_wmma_f32_16x16x32_bf16(
                      false, A, false, Bv, (short)0, acc, false, false);
        }
        if (lane < 16) out[rowoff + 16 * odc + m] = acc[0];
    }
}

extern "C" void kernel_launch(void* const* d_in, const int* in_sizes, int n_in,
                              void* d_out, int out_size, void* d_ws, size_t ws_size,
                              hipStream_t stream) {
    const float* q    = (const float*)d_in[0];
    const float* k    = (const float*)d_in[1];
    const float* v    = (const float*)d_in[2];
    const int*   cols = (const int*)d_in[3];
    float*       out  = (float*)d_out;

    dim3 grid(TOTALQ / WAVES);   // 12288 workgroups
    dim3 block(WAVES * 32);      // 4 wave32 waves per workgroup
    sparse_attn_wmma_kernel<<<grid, block, 0, stream>>>(q, k, v, cols, out);
}
